// RWKV_54769422958622
// MI455X (gfx1250) — matrix operand order repaired and checked by reading, not compiled, not run on hardware
//
#include <hip/hip_runtime.h>
#include <hip/hip_bf16.h>

// ---- problem constants ----
constexpr int kC    = 192;            // embedding dim
constexpr int kB    = 8;              // batch
constexpr int kT    = 16384;          // sequence
constexpr int kRows = kB * kT;        // 131072 rows
constexpr int kNCH  = 128;            // scan chunks
constexpr int kCL   = kT / kNCH;      // 128 steps per chunk
constexpr int kLDST = 200;            // LDS row stride (ushorts), 400B = 25*16B
constexpr int kMT   = 64;             // GEMM M-tile (rows per block)

typedef __bf16 bf16_t;
typedef bf16_t v16bf __attribute__((ext_vector_type(16)));
typedef float  v8f   __attribute__((ext_vector_type(8)));
typedef unsigned short v16us __attribute__((ext_vector_type(16)));

static __device__ __forceinline__ unsigned short f2bf(float f) {
    unsigned int u = __float_as_uint(f);
    u += 0x7fffu + ((u >> 16) & 1u);      // round-to-nearest-even
    return (unsigned short)(u >> 16);
}

static __device__ __forceinline__ v8f wmma_bf16(v16us a, v16us b, v8f c) {
    return __builtin_amdgcn_wmma_f32_16x16x32_bf16(
        false, __builtin_bit_cast(v16bf, a),
        false, __builtin_bit_cast(v16bf, b),
        (short)0, c, false, false);
}

static __device__ __forceinline__ v16us load_afrag(
        const unsigned short* __restrict__ arow, int h) {
    v16us a;
    #pragma unroll
    for (int i = 0; i < 8; ++i) {            // A: M=lane%16, K=16h+i (matches B)
        a[i]     = arow[h * 16 + i];
        a[8 + i] = arow[h * 16 + 8 + i];
    }
    return a;
}

// ---------------------------------------------------------------------------
// Kernel 0: convert the four 192x192 fp32 weight matrices to bf16 (once).
// ---------------------------------------------------------------------------
__global__ void rwkv_convert_weights(const float* __restrict__ wk,
                                     const float* __restrict__ wv,
                                     const float* __restrict__ wr,
                                     const float* __restrict__ wo,
                                     unsigned short* __restrict__ out) {
    const int cc = kC * kC;
    int i = blockIdx.x * blockDim.x + threadIdx.x;
    if (i >= 4 * cc) return;
    int m = i / cc, j = i % cc;
    const float* src = (m == 0) ? wk : (m == 1) ? wv : (m == 2) ? wr : wo;
    out[i] = f2bf(src[j]);
}

// ---------------------------------------------------------------------------
// Kernel 1: fused k/v/sr GEMM.  Block = 384 threads (12 waves), one 64-row
// x-tile in LDS.  Wave w owns output column tile w for all three matrices
// across 4 M-subtiles: each B-fragment load feeds 4 WMMAs -> 72 WMMAs/wave.
// ---------------------------------------------------------------------------
__global__ __launch_bounds__(384) void rwkv_gemm_kvr(
        const float* __restrict__ x,
        const unsigned short* __restrict__ wkb,
        const unsigned short* __restrict__ wvb,
        const unsigned short* __restrict__ wrb,
        float* __restrict__ kout, float* __restrict__ vout,
        float* __restrict__ srout) {
    __shared__ unsigned short xs[kMT * kLDST];      // 25.6 KB

    const int tid = threadIdx.x;
    const int rowbase = blockIdx.x * kMT;

    // stage x tile (64 x 192) as bf16: 32 contiguous elements per thread
    {
        int r  = tid / 6;
        int c0 = (tid % 6) * 32;
        const float* src = x + (size_t)(rowbase + r) * kC + c0;
        unsigned short* dst = xs + r * kLDST + c0;
        #pragma unroll
        for (int i = 0; i < 32; ++i) dst[i] = f2bf(src[i]);
    }
    __syncthreads();

    const int wave = tid >> 5;
    const int lane = tid & 31;
    const int n    = lane & 15;      // also A-row within subtile
    const int h    = lane >> 4;
    const int colbase = wave * 16;

    v8f zero = {0.f,0.f,0.f,0.f,0.f,0.f,0.f,0.f};
    v8f ak[4], av[4], ar[4];
    #pragma unroll
    for (int sub = 0; sub < 4; ++sub) { ak[sub] = zero; av[sub] = zero; ar[sub] = zero; }

    const unsigned short* bk0 = wkb + (size_t)(colbase + n) * kC;
    const unsigned short* bv0 = wvb + (size_t)(colbase + n) * kC;
    const unsigned short* br0 = wrb + (size_t)(colbase + n) * kC;

    #pragma unroll
    for (int s = 0; s < 6; ++s) {
        v16us bk, bv, br;
        const int ko = 32 * s + 16 * h;          // B: N=lane%16, K=16h+i
        #pragma unroll
        for (int i = 0; i < 16; ++i) {
            bk[i] = bk0[ko + i];
            bv[i] = bv0[ko + i];
            br[i] = br0[ko + i];
        }
        #pragma unroll
        for (int sub = 0; sub < 4; ++sub) {
            v16us a = load_afrag(xs + (sub * 16 + n) * kLDST + 32 * s, h);
            ak[sub] = wmma_bf16(a, bk, ak[sub]);
            av[sub] = wmma_bf16(a, bv, av[sub]);
            ar[sub] = wmma_bf16(a, br, ar[sub]);
        }
    }

    #pragma unroll
    for (int sub = 0; sub < 4; ++sub) {
        #pragma unroll
        for (int j = 0; j < 8; ++j) {            // C/D: M=j+8h, N=lane%16
            int m = sub * 16 + j + 8 * h;
            size_t o = (size_t)(rowbase + m) * kC + colbase + n;
            kout[o] = ak[sub][j];
            vout[o] = av[sub][j];
            srout[o] = 1.0f / (1.0f + __expf(-ar[sub][j]));
        }
    }
}

// ---------------------------------------------------------------------------
// WKV chunked parallel scan. True state A = aa*e^pp, B = bb*e^pp.
// Per-step: A' = e^w A + e^k v (linear) -> associative chunk composition.
// ---------------------------------------------------------------------------
// Phase 1: per-chunk transform (P,Q,m): chunk maps A -> e^{wL} A + P.
__global__ __launch_bounds__(192) void rwkv_wkv_phase1(
        const float* __restrict__ kbuf, const float* __restrict__ vbuf,
        const float* __restrict__ sdecay,
        float* __restrict__ spa, float* __restrict__ sqa, float* __restrict__ sm) {
    const int b = blockIdx.x / kNCH, ch = blockIdx.x % kNCH, c = threadIdx.x;
    const float w = sdecay[c] * (1.0f / kT);
    size_t base = ((size_t)b * kT + (size_t)ch * kCL) * kC + c;
    float m = -1e38f, pa = 0.f, qa = 0.f;
    #pragma unroll 4
    for (int j = 0; j < kCL; ++j) {
        float kk = kbuf[base + (size_t)j * kC];
        float vv = vbuf[base + (size_t)j * kC];
        float mw = m + w;
        float mn = fmaxf(mw, kk);
        float e1 = __expf(mw - mn), e2 = __expf(kk - mn);
        pa = e1 * pa + e2 * vv;
        qa = e1 * qa + e2;
        m  = mn;
    }
    size_t idx = (size_t)(b * kNCH + ch) * kC + c;
    spa[idx] = pa; sqa[idx] = qa; sm[idx] = m;
}

// Phase 2: serial scan over 128 chunk transforms; store exclusive prefixes.
__global__ __launch_bounds__(192) void rwkv_wkv_phase2(
        const float* __restrict__ spa, const float* __restrict__ sqa,
        const float* __restrict__ sm, const float* __restrict__ sdecay,
        float* __restrict__ paa, float* __restrict__ pbb, float* __restrict__ ppp) {
    const int b = blockIdx.x, c = threadIdx.x;
    const float w = sdecay[c] * (1.0f / kT);
    const float d = w * (float)kCL;              // log-decay of one chunk
    float aa = 0.f, bb = 0.f, pp = -1e38f;
    for (int ch = 0; ch < kNCH; ++ch) {
        size_t idx = (size_t)(b * kNCH + ch) * kC + c;
        paa[idx] = aa; pbb[idx] = bb; ppp[idx] = pp;   // exclusive prefix
        float pa = spa[idx], qa = sqa[idx], m = sm[idx];
        float pd = pp + d;
        float np = fmaxf(pd, m);
        float e1 = __expf(pd - np), e2 = __expf(m - np);
        aa = e1 * aa + e2 * pa;
        bb = e1 * bb + e2 * qa;
        pp = np;
    }
}

// Phase 3: replay each chunk from its prefix with the exact reference step;
// y overwrites kbuf in place (same element read first by the same thread).
__global__ __launch_bounds__(192) void rwkv_wkv_phase3(
        const float* __restrict__ vbuf,
        const float* __restrict__ sdecay, const float* __restrict__ sfirst,
        const float* __restrict__ paa, const float* __restrict__ pbb,
        const float* __restrict__ ppp,
        float* __restrict__ kybuf) {
    const int b = blockIdx.x / kNCH, ch = blockIdx.x % kNCH, c = threadIdx.x;
    const float w = sdecay[c] * (1.0f / kT);
    const float u = sfirst[c] * (1.0f / kT);
    size_t idx = (size_t)(b * kNCH + ch) * kC + c;
    float aa = paa[idx], bb = pbb[idx], pp = ppp[idx];
    size_t base = ((size_t)b * kT + (size_t)ch * kCL) * kC + c;
    #pragma unroll 2
    for (int j = 0; j < kCL; ++j) {
        size_t o = base + (size_t)j * kC;
        float kk = kybuf[o];
        float vv = vbuf[o];
        float ww = u + kk;
        float q  = fmaxf(pp, ww);
        float e1 = __expf(pp - q), e2 = __expf(ww - q);
        kybuf[o] = (e1 * aa + e2 * vv) / (e1 * bb + e2);   // y
        float ww2 = pp + w;
        float q2  = fmaxf(ww2, kk);
        float e1b = __expf(ww2 - q2), e2b = __expf(kk - q2);
        aa = e1b * aa + e2b * vv;
        bb = e1b * bb + e2b;
        pp = q2;
    }
}

// ---------------------------------------------------------------------------
// Kernel 5: fused LayerNorm + sigmoid-gate + output GEMM (y' @ Wo^T).
// Block = 768 threads (24 waves), 64-row tile: 12 column-tile waves x 2
// subtile groups, 12 WMMAs per wave.
// ---------------------------------------------------------------------------
__global__ __launch_bounds__(768) void rwkv_ln_gemm_out(
        const float* __restrict__ ybuf, const float* __restrict__ srbuf,
        const float* __restrict__ gamma, const float* __restrict__ beta,
        const unsigned short* __restrict__ wob,
        float* __restrict__ out) {
    __shared__ unsigned short at[kMT * kLDST];      // 25.6 KB
    __shared__ float psum[kMT][13];
    __shared__ float psq[kMT][13];
    __shared__ float smu[kMT], srs[kMT];

    const int tid = threadIdx.x;
    const int rowbase = blockIdx.x * kMT;
    const int r = tid / 12, seg = tid % 12, c0 = seg * 16;

    float yv[16], sv[16];
    {
        const float* ys = ybuf  + (size_t)(rowbase + r) * kC + c0;
        const float* ss = srbuf + (size_t)(rowbase + r) * kC + c0;
        float s = 0.f, sq = 0.f;
        #pragma unroll
        for (int i = 0; i < 16; ++i) {
            yv[i] = ys[i]; sv[i] = ss[i];
            s += yv[i]; sq += yv[i] * yv[i];
        }
        psum[r][seg] = s; psq[r][seg] = sq;
    }
    __syncthreads();
    if (tid < kMT) {
        float a = 0.f, b2 = 0.f;
        #pragma unroll
        for (int j = 0; j < 12; ++j) { a += psum[tid][j]; b2 += psq[tid][j]; }
        float mu  = a * (1.0f / kC);
        float var = b2 * (1.0f / kC) - mu * mu;
        smu[tid] = mu;
        srs[tid] = rsqrtf(var + 1e-5f);
    }
    __syncthreads();
    {
        float mu = smu[r], rstd = srs[r];
        unsigned short* dst = at + r * kLDST + c0;
        #pragma unroll
        for (int i = 0; i < 16; ++i) {
            float a = (yv[i] - mu) * rstd * gamma[c0 + i] + beta[c0 + i];
            dst[i] = f2bf(a * sv[i]);
        }
    }
    __syncthreads();

    const int wave = tid >> 5;                   // 0..23
    const int lane = tid & 31;
    const int n = lane & 15, h = lane >> 4;
    const int colbase = (wave % 12) * 16;
    const int sub0    = (wave / 12) * 2;         // subtiles {0,1} or {2,3}

    v8f acc0 = {0.f,0.f,0.f,0.f,0.f,0.f,0.f,0.f};
    v8f acc1 = acc0;
    const unsigned short* b0 = wob + (size_t)(colbase + n) * kC;

    #pragma unroll
    for (int s = 0; s < 6; ++s) {
        v16us bw;
        const int ko = 32 * s + 16 * h;
        #pragma unroll
        for (int i = 0; i < 16; ++i) bw[i] = b0[ko + i];
        v16us a0 = load_afrag(at + (sub0 * 16 + n) * kLDST + 32 * s, h);
        acc0 = wmma_bf16(a0, bw, acc0);
        v16us a1 = load_afrag(at + ((sub0 + 1) * 16 + n) * kLDST + 32 * s, h);
        acc1 = wmma_bf16(a1, bw, acc1);
    }

    #pragma unroll
    for (int j = 0; j < 8; ++j) {
        int m0 = sub0 * 16 + j + 8 * h;
        out[(size_t)(rowbase + m0) * kC + colbase + n] = acc0[j];
        int m1 = (sub0 + 1) * 16 + j + 8 * h;
        out[(size_t)(rowbase + m1) * kC + colbase + n] = acc1[j];
    }
}

// ---------------------------------------------------------------------------
extern "C" void kernel_launch(void* const* d_in, const int* in_sizes, int n_in,
                              void* d_out, int out_size, void* d_ws, size_t ws_size,
                              hipStream_t stream) {
    const float* x      = (const float*)d_in[0];
    const float* Wk     = (const float*)d_in[1];
    const float* Wv     = (const float*)d_in[2];
    const float* Wr     = (const float*)d_in[3];
    const float* Wo     = (const float*)d_in[4];
    const float* sdecay = (const float*)d_in[5];
    const float* sfirst = (const float*)d_in[6];
    const float* gamma  = (const float*)d_in[7];
    const float* beta   = (const float*)d_in[8];

    const size_t cc = (size_t)kC * kC;           // 36864
    const size_t rc = (size_t)kRows * kC;        // 25165824 floats
    const size_t sc = (size_t)kB * kNCH * kC;    // 196608 floats

    char* ws = (char*)d_ws;
    unsigned short* wb = (unsigned short*)ws;    // 4 bf16 weight matrices
    float* kbuf = (float*)(ws + 4 * cc * sizeof(unsigned short)); // 294912 B in
    float* vbuf = kbuf + rc;
    float* srb  = vbuf + rc;
    float* spa  = srb + rc;
    float* sqa  = spa + sc;
    float* sm   = sqa + sc;
    float* paa  = sm  + sc;
    float* pbb  = paa + sc;
    float* ppp  = pbb + sc;
    // total ~307 MB of d_ws

    rwkv_convert_weights<<<(4 * (int)cc + 255) / 256, 256, 0, stream>>>(
        Wk, Wv, Wr, Wo, wb);

    rwkv_gemm_kvr<<<kRows / kMT, 384, 0, stream>>>(
        x, wb, wb + cc, wb + 2 * cc, kbuf, vbuf, srb);

    rwkv_wkv_phase1<<<kB * kNCH, kC, 0, stream>>>(kbuf, vbuf, sdecay, spa, sqa, sm);
    rwkv_wkv_phase2<<<kB, kC, 0, stream>>>(spa, sqa, sm, sdecay, paa, pbb, ppp);
    rwkv_wkv_phase3<<<kB * kNCH, kC, 0, stream>>>(vbuf, sdecay, sfirst,
                                                  paa, pbb, ppp, kbuf);

    rwkv_ln_gemm_out<<<kRows / kMT, 768, 0, stream>>>(
        kbuf, srb, gamma, beta, wb + 3 * cc, (float*)d_out);
}
